// HeadSwitchSelfAttention_15779709845533
// MI455X (gfx1250) — matrix operand order repaired and checked by reading, not compiled, not run on hardware
//
#include <hip/hip_runtime.h>

// ---------------------------------------------------------------------------
// HeadSwitch MoE self-attention for MI455X (gfx1250, wave32, WMMA bf16,
// async global->LDS DMA with ASYNCcnt double buffering)
// B=2, S=1024, D=1024, H=16, DH=64, E=8, K=2
// ---------------------------------------------------------------------------

typedef __bf16 bf16_t;
typedef __attribute__((ext_vector_type(16))) __bf16 v16bf;
typedef __attribute__((ext_vector_type(8)))  __bf16 v8bf;
typedef __attribute__((ext_vector_type(8)))  float  v8f;

#define BB 2
#define SS 1024
#define DD 1024
#define HH 16
#define DH 64
#define EE 8
#define KK 2
#define NTOK (BB*SS)

// ---------------- CDNA5 async copy helpers (ISA 15.18.3, ASYNCcnt) ----------

// 16B global -> LDS direct copy, tracked by ASYNCcnt (in-order completion).
__device__ __forceinline__ void async_copy_b128(const void* g, const void* l) {
  unsigned lds_off = (unsigned)(size_t)l;   // low 32 bits of shared addr = LDS offset
  asm volatile("global_load_async_to_lds_b128 %0, %1, off"
               :: "v"(lds_off), "v"(g)
               : "memory");
}
__device__ __forceinline__ void async_wait_le4() {
  asm volatile("s_wait_asynccnt 4" ::: "memory");
}
__device__ __forceinline__ void async_wait_le0() {
  asm volatile("s_wait_asynccnt 0" ::: "memory");
}

// ---------------- WMMA fragment helpers (layouts per CDNA5 ISA 7.12.2) -----

__device__ __forceinline__ v16bf frag_cat(v8bf lo, v8bf hi) {
  return __builtin_shufflevector(lo, hi, 0,1,2,3,4,5,6,7,8,9,10,11,12,13,14,15);
}

// A fragment: 16x32 bf16, tile row-major [m][k], pitch in halves.
// lanes 0-15:  elems 0-7 -> K=0..7,  elems 8-15 -> K=16..23
// lanes 16-31: elems 0-7 -> K=8..15, elems 8-15 -> K=24..31
__device__ __forceinline__ v16bf load_frag_a(const bf16_t* tile, int pitch) {
  unsigned lane = threadIdx.x & 31u;
  unsigned m  = lane & 15u;
  unsigned kb = (lane & 16u) ? 8u : 0u;
  const bf16_t* p = tile + m * pitch + kb;
  v8bf lo = *(const v8bf*)(p);
  v8bf hi = *(const v8bf*)(p + 16);
  return frag_cat(lo, hi);
}

// B fragment: 32x16 bf16, tile stored as Bt row-major [n][k], pitch in halves.
// elems 0-7 -> K = 8*(lane>=16) + e, elems 8-15 -> K = 16 + 8*(lane>=16) + (e-8), n = lane&15
__device__ __forceinline__ v16bf load_frag_b(const bf16_t* tile, int pitch) {
  unsigned lane = threadIdx.x & 31u;
  unsigned n  = lane & 15u;
  unsigned kb = (lane & 16u) ? 8u : 0u;
  const bf16_t* p = tile + n * pitch + kb;
  v8bf lo = *(const v8bf*)(p);
  v8bf hi = *(const v8bf*)(p + 16);
  return frag_cat(lo, hi);
}

__device__ __forceinline__ v8f wmma_bf16(v16bf a, v16bf b, v8f c) {
  return __builtin_amdgcn_wmma_f32_16x16x32_bf16(false, a, false, b,
                                                 (short)0, c, false, false);
}

// ---------------- elementwise convert / transpose ---------------------------

__global__ void convert_x_kernel(const float* __restrict__ src,
                                 bf16_t* __restrict__ dst, int n) {
  int i = blockIdx.x * blockDim.x + threadIdx.x;
  if (i < n) dst[i] = (bf16_t)src[i];
}

// dst[c][r] = (bf16)src[r][c]; dst pitch = rows
__global__ __launch_bounds__(256)
void transpose_to_bf16(const float* __restrict__ src, bf16_t* __restrict__ dst,
                       int rows, int cols) {
  __shared__ float tile[32][33];
  int c0 = blockIdx.x * 32, r0 = blockIdx.y * 32;
  int tx = threadIdx.x, ty = threadIdx.y;
  #pragma unroll
  for (int k = 0; k < 4; ++k)
    tile[ty + k*8][tx] = src[(size_t)(r0 + ty + k*8) * cols + c0 + tx];
  __syncthreads();
  #pragma unroll
  for (int k = 0; k < 4; ++k)
    dst[(size_t)(c0 + ty + k*8) * rows + r0 + tx] = (bf16_t)tile[tx][ty + k*8];
}

__global__ void zero_counts_kernel(float* __restrict__ counts) {
  counts[threadIdx.x] = 0.0f;
}

// ---------------- generic 64x64-tile bf16 WMMA GEMM -------------------------
// C[m][n] = scale * sum_k A[m][k] * Bt[n][k];  128 threads = 4 waves.
// Double-buffered async global->LDS DMA: 4 b128 copies / thread / k-tile.
template <bool OUT_BF16>
__global__ __launch_bounds__(128)
void gemm64x64(const bf16_t* __restrict__ A, int lda,
               const bf16_t* __restrict__ Bt, int ldb,
               void* __restrict__ Cp, int ldc, int Kd, float scale) {
  __shared__ bf16_t sA[2][64 * 32];
  __shared__ bf16_t sB[2][64 * 32];
  const int m0 = blockIdx.x * 64;
  const int n0 = blockIdx.y * 64;
  const int tid = threadIdx.x;
  const int w = tid >> 5;
  const int rowA = tid >> 1;           // 0..63
  const int colA = (tid & 1) * 16;     // 0 or 16 (halves)

  auto issue_tile = [&](int buf, int k0) {
    const bf16_t* ga = A + (size_t)(m0 + rowA) * lda + k0 + colA;
    const bf16_t* gb = Bt + (size_t)(n0 + rowA) * ldb + k0 + colA;
    async_copy_b128(ga,     &sA[buf][rowA * 32 + colA]);
    async_copy_b128(ga + 8, &sA[buf][rowA * 32 + colA + 8]);
    async_copy_b128(gb,     &sB[buf][rowA * 32 + colA]);
    async_copy_b128(gb + 8, &sB[buf][rowA * 32 + colA + 8]);
  };

  v8f acc[4] = {};
  issue_tile(0, 0);
  for (int k0 = 0; k0 < Kd; k0 += 32) {
    const int cur = (k0 >> 5) & 1;
    const bool more = (k0 + 32) < Kd;
    if (more) issue_tile(cur ^ 1, k0 + 32);   // DMA next tile in background
    if (more) async_wait_le4();               // oldest 4 (current tile) done
    else      async_wait_le0();
    __syncthreads();
    v16bf af = load_frag_a(&sA[cur][w * 16 * 32], 32);
    #pragma unroll
    for (int nt = 0; nt < 4; ++nt) {
      v16bf bf = load_frag_b(&sB[cur][nt * 16 * 32], 32);
      acc[nt] = wmma_bf16(af, bf, acc[nt]);
    }
    __syncthreads();                          // readers done before overwrite
  }
  unsigned lane = tid & 31u;
  int nin = lane & 15;
  int g = (lane >> 4) & 1;
  #pragma unroll
  for (int nt = 0; nt < 4; ++nt)
    #pragma unroll
    for (int r = 0; r < 8; ++r) {
      int m = m0 + w * 16 + r + 8 * g;
      int n = n0 + nt * 16 + nin;
      float v = acc[nt][r] * scale;
      if (OUT_BF16) ((bf16_t*)Cp)[(size_t)m * ldc + n] = (bf16_t)v;
      else          ((float*)Cp)[(size_t)m * ldc + n] = v;
    }
}

// ---------------- scores + causal softmax -----------------------------------
// Block: 256 threads (8 waves). 16 query rows (s-tile) x all 1024 key cols.
// Q is pre-scaled by 1/sqrt(DH). Writes P[bh][s][t] (bf16, normalized rows).
// K tiles (128x64) stream through double-buffered async DMA.
__global__ __launch_bounds__(256)
void scores_softmax_kernel(const bf16_t* __restrict__ Qs,
                           const bf16_t* __restrict__ Kp,
                           bf16_t* __restrict__ P) {
  const int s0 = blockIdx.x * 16;
  const int bh = blockIdx.y;
  const int b = bh >> 4, h = bh & 15;
  __shared__ bf16_t sQ[16 * 64];
  __shared__ bf16_t sK[2][128 * 64];
  __shared__ float pmax[8][16];
  __shared__ float psum[8][16];
  const int tid = threadIdx.x;
  const int w = tid >> 5;
  const unsigned lane = tid & 31u;
  const int ncol = lane & 15;
  const int g = (lane >> 4) & 1;

  const int krow = tid >> 1;           // 0..127
  const int kcol = (tid & 1) * 32;     // 0 or 32 (halves)
  auto issue_chunk = [&](int buf, int c) {
    const bf16_t* gk = Kp + (size_t)(b * SS + c * 128 + krow) * DD + h * 64 + kcol;
    async_copy_b128(gk,      &sK[buf][krow * 64 + kcol]);
    async_copy_b128(gk + 8,  &sK[buf][krow * 64 + kcol + 8]);
    async_copy_b128(gk + 16, &sK[buf][krow * 64 + kcol + 16]);
    async_copy_b128(gk + 24, &sK[buf][krow * 64 + kcol + 24]);
  };

  if (tid < 128) {
    int row = tid >> 3;
    int col = (tid & 7) * 8;
    *(uint4*)&sQ[row * 64 + col] =
        *(const uint4*)(Qs + (size_t)(b * SS + s0 + row) * DD + h * 64 + col);
  }
  issue_chunk(0, 0);
  __syncthreads();
  v16bf a0 = load_frag_a(sQ, 64);
  v16bf a1 = load_frag_a(sQ + 32, 64);

  v8f acc[8] = {};
  for (int c = 0; c < 8; ++c) {
    const int cur = c & 1;
    const bool more = c < 7;
    if (more) issue_chunk(cur ^ 1, c + 1);
    if (more) async_wait_le4();
    else      async_wait_le0();
    __syncthreads();
    const bf16_t* tile = &sK[cur][(w * 16) * 64];
    v16bf b0 = load_frag_b(tile, 64);
    v16bf b1 = load_frag_b(tile + 32, 64);
    acc[c] = wmma_bf16(a0, b0, acc[c]);
    acc[c] = wmma_bf16(a1, b1, acc[c]);
    __syncthreads();
  }

  // causal mask + per-lane row max over all 8 col-chunks
  float m8[8];
  #pragma unroll
  for (int r = 0; r < 8; ++r) m8[r] = -3.0e38f;
  for (int c = 0; c < 8; ++c) {
    int t = c * 128 + w * 16 + ncol;
    #pragma unroll
    for (int r = 0; r < 8; ++r) {
      int s = s0 + r + 8 * g;
      float v = acc[c][r];
      if (t > s) v = -1.0e9f;
      acc[c][r] = v;
      m8[r] = fmaxf(m8[r], v);
    }
  }
  #pragma unroll
  for (int r = 0; r < 8; ++r) {
    float v = m8[r];
    v = fmaxf(v, __shfl_xor(v, 1));
    v = fmaxf(v, __shfl_xor(v, 2));
    v = fmaxf(v, __shfl_xor(v, 4));
    v = fmaxf(v, __shfl_xor(v, 8));
    m8[r] = v;
  }
  if (ncol == 0) {
    #pragma unroll
    for (int r = 0; r < 8; ++r) pmax[w][r + 8 * g] = m8[r];
  }
  __syncthreads();
  float rm[8];
  #pragma unroll
  for (int r = 0; r < 8; ++r) {
    float v = -3.0e38f;
    for (int w2 = 0; w2 < 8; ++w2) v = fmaxf(v, pmax[w2][r + 8 * g]);
    rm[r] = v;
  }
  float l8[8] = {};
  for (int c = 0; c < 8; ++c)
    #pragma unroll
    for (int r = 0; r < 8; ++r) {
      float e = __expf(acc[c][r] - rm[r]);
      acc[c][r] = e;
      l8[r] += e;
    }
  #pragma unroll
  for (int r = 0; r < 8; ++r) {
    float v = l8[r];
    v += __shfl_xor(v, 1);
    v += __shfl_xor(v, 2);
    v += __shfl_xor(v, 4);
    v += __shfl_xor(v, 8);
    l8[r] = v;
  }
  if (ncol == 0) {
    #pragma unroll
    for (int r = 0; r < 8; ++r) psum[w][r + 8 * g] = l8[r];
  }
  __syncthreads();
  float inv[8];
  #pragma unroll
  for (int r = 0; r < 8; ++r) {
    float v = 0.0f;
    for (int w2 = 0; w2 < 8; ++w2) v += psum[w2][r + 8 * g];
    inv[r] = 1.0f / v;
  }
  bf16_t* Pbh = P + (size_t)bh * SS * SS;
  for (int c = 0; c < 8; ++c) {
    int t = c * 128 + w * 16 + ncol;
    #pragma unroll
    for (int r = 0; r < 8; ++r) {
      int s = s0 + r + 8 * g;
      Pbh[(size_t)s * SS + t] = (bf16_t)(acc[c][r] * inv[r]);
    }
  }
}

// ---------------- gating: top-2, softmax, expert histogram ------------------

__global__ void gate_topk_kernel(const float* __restrict__ gates,
                                 float* __restrict__ probs,
                                 int* __restrict__ idx,
                                 float* __restrict__ counts) {
  int gidx = blockIdx.x * blockDim.x + threadIdx.x;   // token*H + h
  if (gidx >= NTOK * HH) return;
  int h = gidx & 15;
  const float* gl = gates + (size_t)(gidx >> 4) * (HH * EE) + h * EE;
  float best = -3.0e38f; int bi = 0;
  #pragma unroll
  for (int e = 0; e < EE; ++e) { float v = gl[e]; if (v > best) { best = v; bi = e; } }
  float sec = -3.0e38f; int si = 0;
  #pragma unroll
  for (int e = 0; e < EE; ++e) {
    if (e == bi) continue;
    float v = gl[e];
    if (v > sec) { sec = v; si = e; }
  }
  float p0 = 1.0f / (1.0f + __expf(sec - best));
  probs[gidx * 2 + 0] = p0;
  probs[gidx * 2 + 1] = 1.0f - p0;
  idx[gidx * 2 + 0] = bi;
  idx[gidx * 2 + 1] = si;
  atomicAdd(&counts[h * EE + bi], 1.0f);
  atomicAdd(&counts[h * EE + si], 1.0f);
}

// ---------------- expert value projection (writes V transposed) -------------
// VpT[bh][kslot*64+e][s] = sum_d x[token, h*64+d] * Wv[h][idx][d][e]
__global__ __launch_bounds__(128)
void vproj_kernel(const float* __restrict__ x,
                  const float* __restrict__ Wv,
                  const int* __restrict__ idx,
                  bf16_t* __restrict__ VpT) {
  int blk = blockIdx.x;                 // token*H + h
  int token = blk >> 4, h = blk & 15;
  int tid = threadIdx.x;
  int kslot = tid >> 6, e = tid & 63;
  int b = token >> 10, s = token & 1023;
  __shared__ float xs[64];
  if (tid < 64) xs[tid] = x[(size_t)token * DD + h * 64 + tid];
  __syncthreads();
  int ex = idx[blk * 2 + kslot];
  const float* W = Wv + ((size_t)(h * EE + ex)) * 64 * 64;
  float acc = 0.0f;
  #pragma unroll 8
  for (int d = 0; d < 64; ++d) acc = fmaf(xs[d], W[d * 64 + e], acc);
  VpT[(((size_t)(b * HH + h)) * 128 + kslot * 64 + e) * SS + s] = (bf16_t)acc;
}

// ---------------- attention combine: attn[t,j] = sum_s P[s,t] * V[s,j] ------
// A = P^T (transposed during LDS staging), Bt = VpT. 128 threads = 4 waves.
__global__ __launch_bounds__(128)
void attn_combine_kernel(const bf16_t* __restrict__ P,
                         const bf16_t* __restrict__ VpT,
                         bf16_t* __restrict__ attn) {
  const int t0 = blockIdx.x * 64;
  const int j0 = blockIdx.y * 64;
  const int bh = blockIdx.z;
  const int b = bh >> 4, h = bh & 15;
  __shared__ bf16_t sPt[64 * 32];
  __shared__ bf16_t sV[64 * 32];
  const int tid = threadIdx.x;
  const int w = tid >> 5;
  v8f acc[4] = {};
  for (int s0 = 0; s0 < SS; s0 += 32) {
    __syncthreads();
    {
      // transpose-stage P tile: sPt[t_local][s_local] = P[s0+s_local][t0+t_local]
      int s_local = tid >> 2;
      int t_base = (tid & 3) * 16;
      const bf16_t* src = P + (size_t)bh * SS * SS + (size_t)(s0 + s_local) * SS + t0 + t_base;
      bf16_t tmp[16];
      *(uint4*)&tmp[0] = *(const uint4*)src;
      *(uint4*)&tmp[8] = *(const uint4*)(src + 8);
      #pragma unroll
      for (int q = 0; q < 16; ++q) sPt[(t_base + q) * 32 + s_local] = tmp[q];
      // direct stage of VpT tile (already [j][s] row-major)
      int jrow = tid >> 1;
      int scol = (tid & 1) * 16;
      const bf16_t* vs = VpT + ((size_t)bh * 128 + j0 + jrow) * SS + s0 + scol;
      *(uint4*)&sV[jrow * 32 + scol]     = *(const uint4*)vs;
      *(uint4*)&sV[jrow * 32 + scol + 8] = *(const uint4*)(vs + 8);
    }
    __syncthreads();
    v16bf af = load_frag_a(sPt + w * 16 * 32, 32);
    #pragma unroll
    for (int nt = 0; nt < 4; ++nt) {
      v16bf bfr = load_frag_b(sV + nt * 16 * 32, 32);
      acc[nt] = wmma_bf16(af, bfr, acc[nt]);
    }
  }
  unsigned lane = tid & 31u;
  int nin = lane & 15;
  int g = (lane >> 4) & 1;
  #pragma unroll
  for (int nt = 0; nt < 4; ++nt)
    #pragma unroll
    for (int r = 0; r < 8; ++r) {
      int t = t0 + w * 16 + r + 8 * g;
      int j = j0 + nt * 16 + nin;
      attn[((size_t)(b * SS + t) * HH + h) * 128 + j] = (bf16_t)acc[nt][r];
    }
}

// ---------------- output projection + gate-weighted sum ---------------------

__global__ __launch_bounds__(64)
void out_kernel(const bf16_t* __restrict__ attn,
                const float* __restrict__ Wo,
                const int* __restrict__ idx,
                const float* __restrict__ probs,
                float* __restrict__ y) {
  int blk = blockIdx.x;                 // token*H + h
  int token = blk >> 4, h = blk & 15;
  int tid = threadIdx.x;                // 64
  __shared__ float sa[128];
  for (int i = tid; i < 128; i += 64) sa[i] = (float)attn[(size_t)blk * 128 + i];
  __syncthreads();
  float out = 0.0f;
  #pragma unroll
  for (int kslot = 0; kslot < KK; ++kslot) {
    int ex = idx[blk * 2 + kslot];
    float pb = probs[blk * 2 + kslot];
    const float* W = Wo + ((size_t)(h * EE + ex)) * 64 * 64;
    float acc = 0.0f;
    #pragma unroll 8
    for (int d = 0; d < 64; ++d) acc = fmaf(sa[kslot * 64 + d], W[d * 64 + tid], acc);
    out = fmaf(pb, acc, out);
  }
  y[(size_t)token * DD + h * 64 + tid] = out;
}

// ---------------- load-balance loss -----------------------------------------

__global__ __launch_bounds__(128)
void lb_loss_kernel(const float* __restrict__ counts, float* __restrict__ loss) {
  __shared__ float red[128];
  int tid = threadIdx.x;
  float ema = counts[tid] * (0.01f / (float)(BB * SS * KK));
  red[tid] = ema;
  __syncthreads();
  for (int off = 64; off > 0; off >>= 1) {
    if (tid < off) red[tid] += red[tid + off];
    __syncthreads();
  }
  float denom = red[0] + 1e-9f;
  __syncthreads();
  float pr = ema / denom;
  red[tid] = pr * pr;
  __syncthreads();
  for (int off = 64; off > 0; off >>= 1) {
    if (tid < off) red[tid] += red[tid + off];
    __syncthreads();
  }
  if (tid == 0) loss[0] = red[0] * (float)(EE * HH);
}

// ---------------------------------------------------------------------------

extern "C" void kernel_launch(void* const* d_in, const int* in_sizes, int n_in,
                              void* d_out, int out_size, void* d_ws, size_t ws_size,
                              hipStream_t stream) {
  const float* x  = (const float*)d_in[0];
  // d_in[1] = mask (recomputed from indices on device)
  const float* Wq = (const float*)d_in[2];
  const float* Wk = (const float*)d_in[3];
  const float* Wv = (const float*)d_in[4];
  const float* Wo = (const float*)d_in[5];
  const float* Wr = (const float*)d_in[6];
  float* y = (float*)d_out;
  float* loss = y + (out_size - 1);

  char* ws = (char*)d_ws;
  size_t off = 0;
  auto alloc = [&](size_t bytes) -> void* {
    void* p = ws + off;
    off = (off + bytes + 255) & ~(size_t)255;
    return p;
  };
  bf16_t* xb    = (bf16_t*)alloc((size_t)NTOK * DD * 2);        // 4 MiB
  bf16_t* WqT   = (bf16_t*)alloc((size_t)DD * DD * 2);          // 2 MiB
  bf16_t* WkT   = (bf16_t*)alloc((size_t)DD * DD * 2);          // 2 MiB
  bf16_t* WrT   = (bf16_t*)alloc((size_t)(HH * EE) * DD * 2);   // 256 KiB
  bf16_t* Qs    = (bf16_t*)alloc((size_t)NTOK * DD * 2);        // 4 MiB
  bf16_t* Kp    = (bf16_t*)alloc((size_t)NTOK * DD * 2);        // 4 MiB
  float*  gates = (float*)alloc((size_t)NTOK * HH * EE * 4);    // 1 MiB
  float*  probs = (float*)alloc((size_t)NTOK * HH * KK * 4);    // 256 KiB
  int*    idx   = (int*)alloc((size_t)NTOK * HH * KK * 4);      // 256 KiB
  float*  counts= (float*)alloc((size_t)HH * EE * 4);           // 512 B
  bf16_t* P     = (bf16_t*)alloc((size_t)BB * HH * SS * SS * 2);// 64 MiB
  bf16_t* VpT   = (bf16_t*)alloc((size_t)BB * HH * 128 * SS * 2);// 8 MiB
  bf16_t* attnb = (bf16_t*)alloc((size_t)NTOK * HH * 128 * 2);  // 8 MiB

  // 1) precision conversion + weight transposes
  convert_x_kernel<<<(NTOK * DD + 255) / 256, 256, 0, stream>>>(x, xb, NTOK * DD);
  transpose_to_bf16<<<dim3(32, 32), dim3(32, 8), 0, stream>>>(Wq, WqT, DD, DD);
  transpose_to_bf16<<<dim3(32, 32), dim3(32, 8), 0, stream>>>(Wk, WkT, DD, DD);
  transpose_to_bf16<<<dim3(4, 32), dim3(32, 8), 0, stream>>>(Wr, WrT, DD, HH * EE);
  zero_counts_kernel<<<1, 128, 0, stream>>>(counts);

  // 2) projections (WMMA): Q scaled by 1/sqrt(DH)=0.125, K, gate logits
  gemm64x64<true><<<dim3(NTOK / 64, DD / 64), 128, 0, stream>>>(
      xb, DD, WqT, DD, Qs, DD, DD, 0.125f);
  gemm64x64<true><<<dim3(NTOK / 64, DD / 64), 128, 0, stream>>>(
      xb, DD, WkT, DD, Kp, DD, DD, 1.0f);
  gemm64x64<false><<<dim3(NTOK / 64, (HH * EE) / 64), 128, 0, stream>>>(
      xb, DD, WrT, DD, gates, HH * EE, DD, 1.0f);

  // 3) gating: top-2 + softmax + histogram
  gate_topk_kernel<<<(NTOK * HH) / 256, 256, 0, stream>>>(gates, probs, idx, counts);

  // 4) scores + causal softmax (WMMA + async K streaming), P[bh][s][t] bf16
  scores_softmax_kernel<<<dim3(SS / 16, BB * HH), 256, 0, stream>>>(Qs, Kp, P);

  // 5) expert value projections -> V^T per (b,h)
  vproj_kernel<<<NTOK * HH, 128, 0, stream>>>(x, Wv, idx, VpT);

  // 6) combine: attn = P^T @ V (WMMA)
  attn_combine_kernel<<<dim3(SS / 64, 128 / 64, BB * HH), 128, 0, stream>>>(P, VpT, attnb);

  // 7) output projection + gate weighting
  out_kernel<<<NTOK * HH, 64, 0, stream>>>(attnb, Wo, idx, probs, y);

  // 8) load-balance loss scalar
  lb_loss_kernel<<<1, 128, 0, stream>>>(counts, loss);
}
